// GraphEncoder_21294447853656
// MI455X (gfx1250) — compile-verified, matrix operand order checked by reading, not run on hardware
//
#include <hip/hip_runtime.h>
#include <hip/hip_bf16.h>
#include <stdint.h>

typedef __attribute__((ext_vector_type(16))) _Float16 v16h;
typedef __attribute__((ext_vector_type(8)))  _Float16 v8h;
typedef __attribute__((ext_vector_type(4)))  _Float16 v4h;
typedef __attribute__((ext_vector_type(8)))  float    v8f;

#define N_NODES 50000
#define N_EDGES 800000
#define IN_DIM  128
#define HIDDEN  256
#define EMB     128
#define HEADS   4

#define GROWS 80   // rows per block (50000 = 625 * 80)
#define MT    5    // 16-row M tiles per wave

// ---------------------------------------------------------------- utilities

__global__ void fill_u32_kernel(uint32_t* p, uint32_t v, size_t n) {
    size_t i = (size_t)blockIdx.x * blockDim.x + threadIdx.x;
    size_t stride = (size_t)gridDim.x * blockDim.x;
    for (; i < n; i += stride) p[i] = v;
}

// n must be a multiple of 4 (always true here)
__global__ void cvt_f32_f16_kernel(const float* __restrict__ src,
                                   _Float16* __restrict__ dst, size_t n4) {
    size_t i = (size_t)blockIdx.x * blockDim.x + threadIdx.x;
    size_t stride = (size_t)gridDim.x * blockDim.x;
    for (; i < n4; i += stride) {
        float4 f = ((const float4*)src)[i];
        v4h h;
        h[0] = (_Float16)f.x; h[1] = (_Float16)f.y;
        h[2] = (_Float16)f.z; h[3] = (_Float16)f.w;
        ((v4h*)dst)[i] = h;
    }
}

// ---------------------------------------------------------------- WMMA GEMM
// out[N,O] = A[N,K](f16) @ W[K,O](f16) + bias[O], f32 accumulate.
// Block: 256 threads = 8 waves. Block tile: 80 rows x 128 cols.
// Wave w owns cols [blockIdx.y*128 + w*16, +16) and 5 M-tiles of 16 rows.
// K chunked by 64 in LDS. A row-major; B TRANSPOSED in LDS so each lane's
// WMMA fragment is two contiguous 16-byte runs -> ds_load_b128.

__global__ __launch_bounds__(256) void wmma_gemm_bias_kernel(
    const _Float16* __restrict__ A, const _Float16* __restrict__ W,
    const float* __restrict__ bias, float* __restrict__ out,
    int K, int O) {
    __shared__ __align__(16) _Float16 As[GROWS * 64];   // 10 KB
    __shared__ __align__(16) _Float16 Bs[128 * 64];     // 16 KB, [col][k]

    const int tid  = threadIdx.x;
    const int lane = tid & 31;
    const int wv   = tid >> 5;
    const int row0 = blockIdx.x * GROWS;
    const int colb = blockIdx.y * 128;

    const int n    = lane & 15;   // A row within tile / B column within tile
    const int half = lane >> 4;

    v8f acc[MT] = {};

    for (int k0 = 0; k0 < K; k0 += 64) {
        // stage A tile (80 x 64 halves) with 128-bit loads
        for (int t = tid; t < GROWS * 8; t += 256) {
            int r = t >> 3, c8 = t & 7;
            *(v8h*)&As[r * 64 + c8 * 8] =
                *(const v8h*)&A[(size_t)(row0 + r) * K + k0 + c8 * 8];
        }
        // stage B tile transposed: global row-major [kk][cc] -> Bs[cc][kk]
        for (int t = tid; t < 64 * 16; t += 256) {
            int kk = t >> 4, c8 = t & 15;
            v8h vv = *(const v8h*)&W[(size_t)(k0 + kk) * O + colb + c8 * 8];
            #pragma unroll
            for (int j = 0; j < 8; ++j) Bs[(c8 * 8 + j) * 64 + kk] = vv[j];
        }
        __syncthreads();

        #pragma unroll
        for (int ks = 0; ks < 64; ks += 32) {
            const _Float16* bp = &Bs[(wv * 16 + n) * 64 + ks + half * 8];
            v8h b0 = *(const v8h*)bp;          // K = ks + half*8 .. +7
            v8h b1 = *(const v8h*)(bp + 16);   // K = ks + 16 + half*8 .. +7
            v16h b = __builtin_shufflevector(b0, b1,
                0,1,2,3,4,5,6,7,8,9,10,11,12,13,14,15);
            #pragma unroll
            for (int mt = 0; mt < MT; ++mt) {
                const _Float16* ap = &As[(mt * 16 + n) * 64 + ks + half * 8];
                v8h a0 = *(const v8h*)ap;
                v8h a1 = *(const v8h*)(ap + 16);
                v16h a = __builtin_shufflevector(a0, a1,
                    0,1,2,3,4,5,6,7,8,9,10,11,12,13,14,15);
                acc[mt] = __builtin_amdgcn_wmma_f32_16x16x32_f16(
                    false, a, false, b, (short)0, acc[mt], false, false);
            }
        }
        __syncthreads();
    }

    const int col = colb + wv * 16 + n;
    const float bv = bias[col];
    #pragma unroll
    for (int mt = 0; mt < MT; ++mt) {
        #pragma unroll
        for (int r = 0; r < 8; ++r) {
            int row = row0 + mt * 16 + half * 8 + r;
            out[(size_t)row * O + col] = acc[mt][r] + bv;
        }
    }
}

// ---------------------------------------------------------------- edge passes

__device__ __forceinline__ void atomicMaxF(float* addr, float val) {
    if (val >= 0.0f) atomicMax((int*)addr, __float_as_int(val));
    else             atomicMin((unsigned int*)addr, __float_as_uint(val));
}

// pass A: alpha[e,h] = q[dst]·(k[src]+we*We)/sqrt(C);  amax[dst,h] = max(...)
__global__ void edge_alpha_kernel(const float* __restrict__ q,
                                  const float* __restrict__ k,
                                  const float* __restrict__ We,
                                  const int* __restrict__ src,
                                  const int* __restrict__ dst,
                                  const float* __restrict__ eattr,
                                  float* __restrict__ alpha,
                                  float* __restrict__ amax,
                                  int O, int C, float rsqC) {
    int t = blockIdx.x * blockDim.x + threadIdx.x;
    if (t >= N_EDGES * HEADS) return;
    int e = t / HEADS, h = t - e * HEADS;
    int s = src[e], d = dst[e];
    float we = eattr[e];
    const float4* qp  = (const float4*)(q  + (size_t)d * O + h * C);
    const float4* kp  = (const float4*)(k  + (size_t)s * O + h * C);
    const float4* Wep = (const float4*)(We + h * C);
    float acc = 0.0f;
    for (int c = 0; c < C / 4; ++c) {
        float4 qv = qp[c], kv = kp[c], wv = Wep[c];
        acc += qv.x * (kv.x + we * wv.x);
        acc += qv.y * (kv.y + we * wv.y);
        acc += qv.z * (kv.z + we * wv.z);
        acc += qv.w * (kv.w + we * wv.w);
    }
    acc *= rsqC;
    alpha[t] = acc;
    atomicMaxF(&amax[d * HEADS + h], acc);
}

// pass B: ea = exp(alpha - amax[dst]);  denom[dst,h] += ea
__global__ void edge_exp_kernel(const int* __restrict__ dst,
                                float* __restrict__ alpha,
                                const float* __restrict__ amax,
                                float* __restrict__ denom) {
    int t = blockIdx.x * blockDim.x + threadIdx.x;
    if (t >= N_EDGES * HEADS) return;
    int e = t / HEADS, h = t - e * HEADS;
    int d = dst[e];
    float ea = __expf(alpha[t] - amax[d * HEADS + h]);
    alpha[t] = ea;
    atomicAdd(&denom[d * HEADS + h], ea);
}

// pass C: agg[dst, idx] += (v[src,idx] + we*We[idx]) * ea / denom
__global__ void edge_scatter_kernel(const float* __restrict__ v,
                                    const float* __restrict__ We,
                                    const int* __restrict__ src,
                                    const int* __restrict__ dst,
                                    const float* __restrict__ eattr,
                                    const float* __restrict__ alpha,
                                    const float* __restrict__ denom,
                                    float* __restrict__ agg,
                                    int O, int logC) {
    int t = blockIdx.x * blockDim.x + threadIdx.x;  // E*O threads (<2^31)
    int e = t / O, idx = t - e * O;
    int h = idx >> logC;
    int s = src[e], d = dst[e];
    float we = eattr[e];
    float a  = alpha[e * HEADS + h] / (denom[d * HEADS + h] + 1e-16f);
    float val = (v[(size_t)s * O + idx] + we * We[idx]) * a;
    atomicAdd(&agg[(size_t)d * O + idx], val);
}

// ---------------------------------------------------------------- node epilogue

__device__ __forceinline__ float block_reduce_sum(float v, float* red, int nt) {
    int i = threadIdx.x;
    red[i] = v;
    __syncthreads();
    for (int s = nt >> 1; s > 0; s >>= 1) {
        if (i < s) red[i] += red[i + s];
        __syncthreads();
    }
    float r = red[0];
    __syncthreads();
    return r;
}

// beta-gated skip + LayerNorm + ReLU + residual. blockDim.x == O, grid == N.
__global__ void node_finish_kernel(const float* __restrict__ agg,
                                   const float* __restrict__ xr,
                                   const float* __restrict__ xprev,
                                   const float* __restrict__ Wb,
                                   const float* __restrict__ ln_g,
                                   const float* __restrict__ ln_b,
                                   float* __restrict__ xnext,
                                   int O, int doRelu, int doResid) {
    __shared__ float red[256];
    const int n = blockIdx.x, i = threadIdx.x;
    const float ov = agg[(size_t)n * O + i];
    const float xv = xr[(size_t)n * O + i];

    float s  = ov * Wb[i] + xv * Wb[O + i] + (ov - xv) * Wb[2 * O + i];
    float bt = block_reduce_sum(s, red, O);
    float beta = 1.0f / (1.0f + __expf(-bt));
    float y = beta * xv + (1.0f - beta) * ov;

    float mu = block_reduce_sum(y, red, O) / (float)O;
    float dv = y - mu;
    float var = block_reduce_sum(dv * dv, red, O) / (float)O;
    float yn = dv * rsqrtf(var + 1e-5f) * ln_g[i] + ln_b[i];
    if (doRelu) yn = fmaxf(yn, 0.0f);
    if (doResid) yn += xprev[(size_t)n * O + i];
    xnext[(size_t)n * O + i] = yn;
}

// ---------------------------------------------------------------- host driver

extern "C" void kernel_launch(void* const* d_in, const int* in_sizes, int n_in,
                              void* d_out, int out_size, void* d_ws, size_t ws_size,
                              hipStream_t stream) {
    const float* x0    = (const float*)d_in[0];
    const int*   ei    = (const int*)d_in[1];          // [2, E]
    const float* eattr = (const float*)d_in[2];        // [E, 1]
    const int* src = ei;
    const int* dst = ei + N_EDGES;

    // Param flatten order: insertion vs JAX-sorted keys.
    int IW_Q, IB_Q, IW_K, IB_K, IW_V, IB_V, I_WE, IW_S, IB_S, I_WB, I_LG, I_LB;
    if (in_sizes[3] == IN_DIM * HIDDEN) {   // insertion: Wq first (128*256)
        IW_Q=0; IB_Q=1; IW_K=2; IB_K=3; IW_V=4; IB_V=5;
        I_WE=6; IW_S=7; IB_S=8; I_WB=9; I_LG=10; I_LB=11;
    } else {                                // sorted: We,Wbeta,Wk,Wq,Wskip,Wv,bk,bq,bskip,bv,ln_b,ln_g
        I_WE=0; I_WB=1; IW_K=2; IW_Q=3; IW_S=4; IW_V=5;
        IB_K=6; IB_Q=7; IB_S=8; IB_V=9; I_LB=10; I_LG=11;
    }

    // workspace carve-up
    char* w = (char*)d_ws;
    size_t off = 0;
    auto carve = [&](size_t bytes) -> void* {
        void* p = w + off;
        off = (off + bytes + 255) & ~(size_t)255;
        return p;
    };
    _Float16* xh  = (_Float16*)carve((size_t)N_NODES * 256 * 2);
    _Float16* whq = (_Float16*)carve((size_t)HIDDEN * HIDDEN * 2);
    _Float16* whk = (_Float16*)carve((size_t)HIDDEN * HIDDEN * 2);
    _Float16* whv = (_Float16*)carve((size_t)HIDDEN * HIDDEN * 2);
    _Float16* whs = (_Float16*)carve((size_t)HIDDEN * HIDDEN * 2);
    float* qb    = (float*)carve((size_t)N_NODES * 256 * 4);
    float* kb    = (float*)carve((size_t)N_NODES * 256 * 4);
    float* vb    = (float*)carve((size_t)N_NODES * 256 * 4);
    float* xrb   = (float*)carve((size_t)N_NODES * 256 * 4);
    float* aggb  = (float*)carve((size_t)N_NODES * 256 * 4);
    float* xbuf0 = (float*)carve((size_t)N_NODES * 256 * 4);
    float* xbuf1 = (float*)carve((size_t)N_NODES * 256 * 4);
    float* alphab = (float*)carve((size_t)N_EDGES * HEADS * 4);
    float* amaxb  = (float*)carve((size_t)N_NODES * HEADS * 4);
    float* denomb = (float*)carve((size_t)N_NODES * HEADS * 4);
    (void)ws_size; (void)n_in; (void)out_size;

    const int inD[4]  = {IN_DIM, HIDDEN, HIDDEN, HIDDEN};
    const int outD[4] = {HIDDEN, HIDDEN, HIDDEN, EMB};
    const float* xcur = x0;

    for (int L = 0; L < 4; ++L) {
        const int base = 3 + 12 * L;
        const int K = inD[L], O = outD[L];
        const int C = O / HEADS;
        const int logC = (C == 64) ? 6 : 5;
        const float rsqC = rsqrtf((float)C);
        const float* Wq = (const float*)d_in[base + IW_Q];
        const float* bq = (const float*)d_in[base + IB_Q];
        const float* Wk = (const float*)d_in[base + IW_K];
        const float* bk = (const float*)d_in[base + IB_K];
        const float* Wv = (const float*)d_in[base + IW_V];
        const float* bv = (const float*)d_in[base + IB_V];
        const float* We = (const float*)d_in[base + I_WE];
        const float* Ws = (const float*)d_in[base + IW_S];
        const float* bs = (const float*)d_in[base + IB_S];
        const float* Wb = (const float*)d_in[base + I_WB];
        const float* lg = (const float*)d_in[base + I_LG];
        const float* lb = (const float*)d_in[base + I_LB];

        // f32 -> f16 conversions (vectorized x4)
        {
            size_t n4 = (size_t)N_NODES * K / 4;
            cvt_f32_f16_kernel<<<(unsigned)((n4 + 255) / 256), 256, 0, stream>>>(xcur, xh, n4);
            size_t nw4 = (size_t)K * O / 4;
            unsigned gb = (unsigned)((nw4 + 255) / 256);
            cvt_f32_f16_kernel<<<gb, 256, 0, stream>>>(Wq, whq, nw4);
            cvt_f32_f16_kernel<<<gb, 256, 0, stream>>>(Wk, whk, nw4);
            cvt_f32_f16_kernel<<<gb, 256, 0, stream>>>(Wv, whv, nw4);
            cvt_f32_f16_kernel<<<gb, 256, 0, stream>>>(Ws, whs, nw4);
        }

        // WMMA GEMMs: q, k, v, x_r
        {
            dim3 grid(N_NODES / GROWS, O / 128);
            wmma_gemm_bias_kernel<<<grid, 256, 0, stream>>>(xh, whq, bq, qb, K, O);
            wmma_gemm_bias_kernel<<<grid, 256, 0, stream>>>(xh, whk, bk, kb, K, O);
            wmma_gemm_bias_kernel<<<grid, 256, 0, stream>>>(xh, whv, bv, vb, K, O);
            wmma_gemm_bias_kernel<<<grid, 256, 0, stream>>>(xh, whs, bs, xrb, K, O);
        }

        // init segment buffers
        {
            size_t nh = (size_t)N_NODES * HEADS;
            fill_u32_kernel<<<(unsigned)((nh + 255) / 256), 256, 0, stream>>>(
                (uint32_t*)amaxb, 0xFF800000u, nh);              // -inf
            fill_u32_kernel<<<(unsigned)((nh + 255) / 256), 256, 0, stream>>>(
                (uint32_t*)denomb, 0u, nh);
            size_t no = (size_t)N_NODES * O;
            fill_u32_kernel<<<(unsigned)((no + 255) / 256), 256, 0, stream>>>(
                (uint32_t*)aggb, 0u, no);
        }

        // edge attention
        {
            int eh = N_EDGES * HEADS;
            edge_alpha_kernel<<<(eh + 255) / 256, 256, 0, stream>>>(
                qb, kb, We, src, dst, eattr, alphab, amaxb, O, C, rsqC);
            edge_exp_kernel<<<(eh + 255) / 256, 256, 0, stream>>>(
                dst, alphab, amaxb, denomb);
            int eo = N_EDGES * O;   // <= 204.8M, fits int
            edge_scatter_kernel<<<eo / 256, 256, 0, stream>>>(
                vb, We, src, dst, eattr, alphab, denomb, aggb, O, logC);
        }

        // node epilogue
        {
            int doRelu  = (L != 3);
            int doResid = (K == O);
            float* xnext = (L == 3) ? (float*)d_out
                                    : ((L & 1) ? xbuf1 : xbuf0);
            node_finish_kernel<<<N_NODES, O, 0, stream>>>(
                aggb, xrb, xcur, Wb, lg, lb, xnext, O, doRelu, doResid);
            xcur = xnext;
        }
    }
}
